// ExpertChoiceRouter_42691974922247
// MI455X (gfx1250) — matrix operand order repaired and checked by reading, not compiled, not run on hardware
//
#include <hip/hip_runtime.h>

// ExpertChoiceRouter for MI455X (gfx1250, wave32, WMMA).
//   x: [4,8192,4096] f32, W: [64,4096] f32
//   out = concat(mask[4,8192,64], probs[4,8192,64], logits[4,8192,64]) f32
//
// Pipeline (all on `stream`):
//   k0: W f32 -> f16 into d_ws (512 KB)            (B operand served from L2)
//   k1: fused GEMM (v_wmma_f32_16x16x32_f16, f32 accum) + softmax epilogue
//       M=32 tokens/wave x all 64 experts: 16 b128 loads : 8 WMMA per K-iter
//   k2: zero the mask region
//   k3: per-(b,e) top-256 selection (bitwise float binary search) -> mask ch.0

#define EMBED   4096
#define NEXP    64
#define SEQ     8192
#define BATCH   4
#define TOKENS  (BATCH * SEQ)     // 32768
#define CAP     256
#define OUTPLANE ((size_t)TOKENS * NEXP)   // 2,097,152 floats per output

typedef __attribute__((ext_vector_type(16))) _Float16 v16h;
typedef __attribute__((ext_vector_type(4)))  _Float16 v4h;
typedef __attribute__((ext_vector_type(8)))  float    v8f;
typedef __attribute__((ext_vector_type(4)))  float    v4f;

// ---------------------------------------------------------------- k0: W->f16
__global__ __launch_bounds__(256) void convert_w_f16(const float* __restrict__ W,
                                                     _Float16* __restrict__ w16) {
  int i = (blockIdx.x * 256 + threadIdx.x) * 4;       // 64*4096 = 262144 elems
  v4f f = *(const v4f*)(W + i);
  v4h h;
  h[0] = (_Float16)f[0]; h[1] = (_Float16)f[1];
  h[2] = (_Float16)f[2]; h[3] = (_Float16)f[3];
  *(v4h*)(w16 + i) = h;
}

// ---------------------------------------------- k1: WMMA GEMM + fused softmax
// Grid: 128 blocks x 256 threads (8 waves). Each wave: 32 tokens x 64 experts.
__global__ __launch_bounds__(256) void router_gemm_softmax(
    const float* __restrict__ x, const _Float16* __restrict__ w16,
    float* __restrict__ out_probs, float* __restrict__ out_logits) {
  const int wave  = threadIdx.x >> 5;
  const int lane  = threadIdx.x & 31;
  const int tile  = blockIdx.x * 8 + wave;            // 0..1023
  const int tok0  = tile * 32;
  const int m     = lane & 15;                        // A row / C column
  const int khalf = lane >> 4;                        // K half-group select

  // A: lane needs row m, 16 contiguous K at k0 + khalf*16
  const float* arow0 = x + (size_t)(tok0 + m) * EMBED + khalf * 16;
  const float* arow1 = arow0 + (size_t)16 * EMBED;    // second M-tile
  // B: lane needs W row (m + 16t), 16 contiguous K at k0 + khalf*16
  const _Float16* brow = w16 + (size_t)m * EMBED + khalf * 16;

  v8f c00 = {}, c01 = {}, c02 = {}, c03 = {};         // M-tile 0, N-tiles 0..3
  v8f c10 = {}, c11 = {}, c12 = {}, c13 = {};         // M-tile 1, N-tiles 0..3

  for (int k0 = 0; k0 < EMBED; k0 += 32) {
    const float* ap0 = arow0 + k0;
    const float* ap1 = arow1 + k0;
    __builtin_prefetch(ap0 + 64, 0, 3);               // WGP-scope prefetch ahead
    __builtin_prefetch(ap1 + 64, 0, 3);

    // B operand: 4 N-tiles, shared by both M-tiles this iteration.
    const _Float16* bp = brow + k0;
    v16h b0 = *(const v16h*)(bp + (size_t) 0 * EMBED);
    v16h b1 = *(const v16h*)(bp + (size_t)16 * EMBED);
    v16h b2 = *(const v16h*)(bp + (size_t)32 * EMBED);
    v16h b3 = *(const v16h*)(bp + (size_t)48 * EMBED);

    v4f f0 = *(const v4f*)(ap0 +  0);
    v4f f1 = *(const v4f*)(ap0 +  4);
    v4f f2 = *(const v4f*)(ap0 +  8);
    v4f f3 = *(const v4f*)(ap0 + 12);
    v4f g0 = *(const v4f*)(ap1 +  0);
    v4f g1 = *(const v4f*)(ap1 +  4);
    v4f g2 = *(const v4f*)(ap1 +  8);
    v4f g3 = *(const v4f*)(ap1 + 12);

    v16h a0, a1;
#pragma unroll
    for (int i = 0; i < 4; ++i) {
      a0[i]      = (_Float16)f0[i];
      a0[4 + i]  = (_Float16)f1[i];
      a0[8 + i]  = (_Float16)f2[i];
      a0[12 + i] = (_Float16)f3[i];
      a1[i]      = (_Float16)g0[i];
      a1[4 + i]  = (_Float16)g1[i];
      a1[8 + i]  = (_Float16)g2[i];
      a1[12 + i] = (_Float16)g3[i];
    }

    c00 = __builtin_amdgcn_wmma_f32_16x16x32_f16(false, a0, false, b0, (short)0, c00, false, false);
    c01 = __builtin_amdgcn_wmma_f32_16x16x32_f16(false, a0, false, b1, (short)0, c01, false, false);
    c02 = __builtin_amdgcn_wmma_f32_16x16x32_f16(false, a0, false, b2, (short)0, c02, false, false);
    c03 = __builtin_amdgcn_wmma_f32_16x16x32_f16(false, a0, false, b3, (short)0, c03, false, false);
    c10 = __builtin_amdgcn_wmma_f32_16x16x32_f16(false, a1, false, b0, (short)0, c10, false, false);
    c11 = __builtin_amdgcn_wmma_f32_16x16x32_f16(false, a1, false, b1, (short)0, c11, false, false);
    c12 = __builtin_amdgcn_wmma_f32_16x16x32_f16(false, a1, false, b2, (short)0, c12, false, false);
    c13 = __builtin_amdgcn_wmma_f32_16x16x32_f16(false, a1, false, b3, (short)0, c13, false, false);
  }

  // Epilogue: softmax over 64 experts per token.
  // C layout: VGPR r, lanes 0-15 -> M=r, lanes 16-31 -> M=r+8; N = lane&15 (+16t).
#pragma unroll
  for (int mt = 0; mt < 2; ++mt) {
    const v8f& d0 = mt ? c10 : c00;
    const v8f& d1 = mt ? c11 : c01;
    const v8f& d2 = mt ? c12 : c02;
    const v8f& d3 = mt ? c13 : c03;
#pragma unroll
    for (int r = 0; r < 8; ++r) {
      const int tok = tok0 + mt * 16 + r + 8 * khalf;
      float v0 = d0[r], v1 = d1[r], v2 = d2[r], v3 = d3[r];
      float mx = fmaxf(fmaxf(v0, v1), fmaxf(v2, v3));
#pragma unroll
      for (int off = 1; off < 16; off <<= 1)
        mx = fmaxf(mx, __shfl_xor(mx, off, 32));      // stays within 16-lane half
      float e0 = __expf(v0 - mx), e1 = __expf(v1 - mx);
      float e2 = __expf(v2 - mx), e3 = __expf(v3 - mx);
      float s = e0 + e1 + e2 + e3;
#pragma unroll
      for (int off = 1; off < 16; off <<= 1)
        s += __shfl_xor(s, off, 32);
      float inv = 1.0f / s;
      size_t base = (size_t)tok * NEXP + m;
      out_logits[base +  0] = v0; out_logits[base + 16] = v1;
      out_logits[base + 32] = v2; out_logits[base + 48] = v3;
      out_probs [base +  0] = e0 * inv; out_probs [base + 16] = e1 * inv;
      out_probs [base + 32] = e2 * inv; out_probs [base + 48] = e3 * inv;
    }
  }
}

// ----------------------------------------------------------- k2: zero mask
__global__ __launch_bounds__(256) void zero_mask(float* __restrict__ p, int n4) {
  int i = blockIdx.x * 256 + threadIdx.x;
  if (i < n4) ((float4*)p)[i] = make_float4(0.f, 0.f, 0.f, 0.f);
}

// ------------------------------------------- k3: top-256 per (b,e) -> mask ch0
// One block (256 threads) per (b,e). Thread tid owns contiguous tokens
// [tid*32, tid*32+32). Bitwise binary search on float bits (probs >= 0).
__global__ __launch_bounds__(256) void topk_mask(const float* __restrict__ probs,
                                                 float* __restrict__ mask) {
  __shared__ int scnt[8];
  __shared__ int stie[256];
  __shared__ int red;

  const int be   = blockIdx.x;          // b*64 + e
  const int b    = be >> 6;
  const int e    = be & 63;
  const int tid  = threadIdx.x;
  const int lane = tid & 31;
  const int wv   = tid >> 5;
  const int s0   = tid * 32;

  const float* pb = probs + (size_t)b * SEQ * NEXP + e;
  float v[32];
#pragma unroll
  for (int i = 0; i < 32; ++i) v[i] = pb[(size_t)(s0 + i) * NEXP];

  // Find V = 256th largest value: largest t with count(p >= t) >= CAP.
  unsigned tv = 0;
  for (int bit = 30; bit >= 0; --bit) {
    const unsigned cand = tv | (1u << bit);
    const float cf = __uint_as_float(cand);
    int cnt = 0;
#pragma unroll
    for (int i = 0; i < 32; ++i) cnt += (v[i] >= cf) ? 1 : 0;
#pragma unroll
    for (int off = 1; off < 32; off <<= 1) cnt += __shfl_xor(cnt, off, 32);
    if (lane == 0) scnt[wv] = cnt;
    __syncthreads();
    if (tid == 0) {
      int t = 0;
      for (int w = 0; w < 8; ++w) t += scnt[w];
      red = t;
    }
    __syncthreads();
    if (red >= CAP) tv = cand;
  }
  const float Vf = __uint_as_float(tv);

  // count strictly-greater -> how many ties to admit (lowest index first)
  int cg = 0, myties = 0;
#pragma unroll
  for (int i = 0; i < 32; ++i) {
    cg     += (v[i] >  Vf) ? 1 : 0;
    myties += (v[i] == Vf) ? 1 : 0;
  }
#pragma unroll
  for (int off = 1; off < 32; off <<= 1) cg += __shfl_xor(cg, off, 32);
  if (lane == 0) scnt[wv] = cg;
  stie[tid] = myties;
  __syncthreads();
  if (tid == 0) {
    int t = 0;
    for (int w = 0; w < 8; ++w) t += scnt[w];
    red = t;
  }
  __syncthreads();
  int need = CAP - red;                 // ties admitted in index order
  int rank = 0;
  for (int t = 0; t < tid; ++t) rank += stie[t];  // exclusive prefix of ties

  float* mrow = mask + (size_t)b * SEQ * NEXP;    // channel 0 only
#pragma unroll
  for (int i = 0; i < 32; ++i) {
    bool sel = (v[i] > Vf);
    if (v[i] == Vf) { sel = sel || (rank < need); ++rank; }
    if (sel) mrow[(size_t)(s0 + i) * NEXP] = 1.0f;  // idempotent 1.0 store
  }
}

// ------------------------------------------------------------------- launcher
extern "C" void kernel_launch(void* const* d_in, const int* in_sizes, int n_in,
                              void* d_out, int out_size, void* d_ws, size_t ws_size,
                              hipStream_t stream) {
  const float* x = (const float*)d_in[0];
  const float* W = (const float*)d_in[1];
  float* out    = (float*)d_out;
  float* mask   = out;                  // [4,8192,64]
  float* probs  = out + OUTPLANE;       // [4,8192,64]
  float* logits = out + 2 * OUTPLANE;   // [4,8192,64]
  _Float16* w16 = (_Float16*)d_ws;      // 64*4096*2 = 512 KB scratch

  convert_w_f16      <<<256, 256, 0, stream>>>(W, w16);
  router_gemm_softmax<<<128, 256, 0, stream>>>(x, w16, probs, logits);
  zero_mask          <<<(int)(OUTPLANE / 4 + 255) / 256, 256, 0, stream>>>(mask, (int)(OUTPLANE / 4));
  topk_mask          <<<BATCH * NEXP, 256, 0, stream>>>(probs, mask);
}
